// TtRobertaSelfAttention_57406532879022
// MI455X (gfx1250) — matrix operand order repaired and checked by reading, not compiled, not run on hardware
//
#include <hip/hip_runtime.h>
#include <hip/hip_bf16.h>

#define BB 4
#define SS 2048
#define DD 1024
#define HH 16
#define HD 64

typedef __bf16 bf16_t;
typedef __attribute__((ext_vector_type(16))) __bf16 v16bf;
typedef __attribute__((ext_vector_type(8)))  __bf16 v8bf;
typedef __attribute__((ext_vector_type(4)))  __bf16 v4bf;
typedef __attribute__((ext_vector_type(8)))  float  v8f;
typedef __attribute__((ext_vector_type(4)))  int    v4i;

typedef __attribute__((address_space(1))) v4i* gv4i_p;   // global int4*
typedef __attribute__((address_space(3))) v4i* sv4i_p;   // LDS int4*
typedef __attribute__((address_space(3))) void svoid_t;

__device__ __forceinline__ v16bf load16(const bf16_t* p) { return *(const v16bf*)p; }
__device__ __forceinline__ v8bf  load8 (const bf16_t* p) { return *(const v8bf*)p;  }
__device__ __forceinline__ bf16_t to_bf16(float f) { return (bf16_t)f; }

__device__ __forceinline__ v16bf combine8(v8bf lo, v8bf hi) {
  v16bf r;
#pragma unroll
  for (int i = 0; i < 8; ++i) { r[i] = lo[i]; r[i + 8] = hi[i]; }
  return r;
}

// --- CDNA5 async copy: global -> LDS (tracked by ASYNCcnt) ---------------
__device__ __forceinline__ void async_copy_b128(const bf16_t* g, bf16_t* l) {
#if __has_builtin(__builtin_amdgcn_global_load_async_to_lds_b128)
  __builtin_amdgcn_global_load_async_to_lds_b128((gv4i_p)g, (sv4i_p)l, 0, 0);
#else
  unsigned lds_off = (unsigned)(__UINTPTR_TYPE__)(svoid_t*)l;
  asm volatile("global_load_async_to_lds_b128 %0, %1, off"
               :: "v"(lds_off), "v"((unsigned long long)(__UINTPTR_TYPE__)g)
               : "memory");
#endif
}

template <int N>
__device__ __forceinline__ void wait_asynccnt() {
  asm volatile("s_wait_asynccnt %0" :: "i"(N) : "memory");
}

// ---------------------------------------------------------------------------
// fp32 -> bf16 conversion, 4 elements per thread
// ---------------------------------------------------------------------------
__global__ void cvt_f32_bf16_kernel(const float4* __restrict__ in,
                                    v4bf* __restrict__ out, int n4) {
  int i = blockIdx.x * blockDim.x + threadIdx.x;
  if (i < n4) {
    float4 f = in[i];
    v4bf o;
    o[0] = to_bf16(f.x); o[1] = to_bf16(f.y);
    o[2] = to_bf16(f.z); o[3] = to_bf16(f.w);
    out[i] = o;
  }
}

// ---------------------------------------------------------------------------
// QKV projection GEMM: out = X @ W^T + b, X:[M=B*S,K=D] bf16, W:[N=D,K=D] bf16
// One wave computes a 32(M) x 64(N) strip: 8 accumulators, A reused x4,
// B reused x2 -> 12 b128 loads per 8 WMMAs. blockIdx.y == head.
// transpose_out=0 -> [B,H,S,HD], =1 -> Vt [B,H,HD,S].
// ---------------------------------------------------------------------------
__global__ void __launch_bounds__(32)
qkv_gemm_kernel(const bf16_t* __restrict__ X, const bf16_t* __restrict__ W,
                const float* __restrict__ bias, bf16_t* __restrict__ out,
                int transpose_out) {
  const int lane = threadIdx.x & 31;
  const int half = lane >> 4;
  const int c    = lane & 15;
  const int m0   = blockIdx.x * 32;
  const int h    = blockIdx.y;        // N tile of 64 == head index
  const int n0   = h * 64;

  v8f acc[2][4];
#pragma unroll
  for (int mt = 0; mt < 2; ++mt)
#pragma unroll
    for (int nt = 0; nt < 4; ++nt) acc[mt][nt] = v8f{};

  const bf16_t* Arow0 = X + (size_t)(m0 + c) * DD;
  const bf16_t* Arow1 = X + (size_t)(m0 + 16 + c) * DD;
#pragma unroll 2
  for (int k0 = 0; k0 < DD; k0 += 32) {
    v16bf a0 = load16(Arow0 + k0 + half * 16);
    v16bf a1 = load16(Arow1 + k0 + half * 16);
    const bf16_t* Wk = W + k0 + half * 16;
#pragma unroll
    for (int nt = 0; nt < 4; ++nt) {
      v16bf b = load16(Wk + (size_t)(n0 + nt * 16 + c) * DD);
      acc[0][nt] = __builtin_amdgcn_wmma_f32_16x16x32_bf16(false, a0, false, b, (short)0, acc[0][nt], false, false);
      acc[1][nt] = __builtin_amdgcn_wmma_f32_16x16x32_bf16(false, a1, false, b, (short)0, acc[1][nt], false, false);
    }
  }

  const int b  = m0 / SS;
  const int s0 = m0 % SS;

#pragma unroll
  for (int mt = 0; mt < 2; ++mt) {
    const int sbase = s0 + mt * 16;
    if (!transpose_out) {
      bf16_t* dst = out + ((size_t)(b * HH + h) * SS) * HD;
#pragma unroll
      for (int nt = 0; nt < 4; ++nt) {
        const int hd = nt * 16 + c;
        const float bv = bias[n0 + hd];
#pragma unroll
        for (int j = 0; j < 8; ++j)
          dst[(size_t)(sbase + j + 8 * half) * HD + hd] = to_bf16(acc[mt][nt][j] + bv);
      }
    } else {
      // Vt [B,H,HD,S]: 8 j-values contiguous in s -> packed 16B store
#pragma unroll
      for (int nt = 0; nt < 4; ++nt) {
        const int hd = nt * 16 + c;
        const float bv = bias[n0 + hd];
        v8bf o;
#pragma unroll
        for (int j = 0; j < 8; ++j) o[j] = to_bf16(acc[mt][nt][j] + bv);
        bf16_t* dst = out + ((size_t)(b * HH + h) * HD + hd) * SS + sbase + 8 * half;
        *(v8bf*)dst = o;
      }
    }
  }
}

// ---------------------------------------------------------------------------
// Flash attention with async double-buffered K/V staging in LDS.
// Block = 4 waves sharing one (b,h); each wave owns a 16-query tile.
// Per 32-key iteration the block async-copies the NEXT K tile (32x64, 4KB,
// contiguous) and V tile (64x32 from Vt, 4KB) into LDS, waits ASYNCcnt for
// the CURRENT tile, then runs 8 WMMAs fed from LDS. Online softmax in fp32.
// ---------------------------------------------------------------------------
__global__ void __launch_bounds__(128)
flash_attn_kernel(const bf16_t* __restrict__ Q, const bf16_t* __restrict__ K,
                  const bf16_t* __restrict__ Vt, const float* __restrict__ mask,
                  float* __restrict__ out) {
  __shared__ __align__(16) bf16_t ldsK[2][32][64];   // [buf][key_local][d]
  __shared__ __align__(16) bf16_t ldsV[2][64][32];   // [buf][d][key_local]
  __shared__ __align__(16) bf16_t pl[4][16][32];     // per-wave P relay

  const int tid  = threadIdx.x;
  const int wid  = tid >> 5;
  const int lane = tid & 31;
  const int half = lane >> 4;
  const int c    = lane & 15;
  const int bh   = blockIdx.y;
  const int b    = bh / HH;
  const int h    = bh % HH;
  const int q0   = (blockIdx.x * 4 + wid) * 16;

  const bf16_t* Qh = Q  + (size_t)bh * SS * HD;
  const bf16_t* Kh = K  + (size_t)bh * SS * HD;
  const bf16_t* Vh = Vt + (size_t)bh * HD * SS;
  const float*  mb = mask + (size_t)b * SS;

  // Q A-operands (d=0..31, d=32..63) held in VGPRs for the whole pass
  const bf16_t* qrow = Qh + (size_t)(q0 + c) * HD;
  v16bf qa0 = load16(qrow +      half * 16);
  v16bf qa1 = load16(qrow + 32 + half * 16);

  v8f acc[4] = {v8f{}, v8f{}, v8f{}, v8f{}};
  float mrow[8], lrow[8];
#pragma unroll
  for (int j = 0; j < 8; ++j) { mrow[j] = -3.0e38f; lrow[j] = 0.0f; }

  const float scale = 0.125f;   // 1/sqrt(64)

  // cooperative async stage of one 32-key K/V tile pair (4 x b128 per thread)
  auto stage_kv = [&](int buf, int kt) {
#pragma unroll
    for (int t = 0; t < 2; ++t) {
      const int idx = tid + t * 128;               // 0..255 x 16B chunks
      // K tile: contiguous 4KB block
      async_copy_b128(Kh + (size_t)kt * HD + idx * 8,
                      &ldsK[buf][0][0] + idx * 8);
      // V tile: 64 rows (d), 32 cols (key), row stride S in global
      const int r = idx >> 2, sub = idx & 3;
      async_copy_b128(Vh + (size_t)r * SS + kt + sub * 8,
                      &ldsV[buf][r][sub * 8]);
    }
  };

  stage_kv(0, 0);   // prologue: tile 0 in flight

  for (int kt = 0; kt < SS; kt += 32) {
    const int buf = (kt >> 5) & 1;
    if (kt + 32 < SS) {
      stage_kv(buf ^ 1, kt + 32);    // prefetch next tile into other buffer
      wait_asynccnt<4>();            // 4 newest (next tile) may stay in flight
    } else {
      wait_asynccnt<0>();
    }
    __syncthreads();                 // all waves' portions of current tile landed

    // --- scores: two 16x16 tiles from LDS K ---
    v8f s0 = {}, s1 = {};
    {
      const bf16_t* kr0 = &ldsK[buf][c][0];
      const bf16_t* kr1 = &ldsK[buf][16 + c][0];
      s0 = __builtin_amdgcn_wmma_f32_16x16x32_bf16(false, qa0, false, load16(kr0 + half * 16),      (short)0, s0, false, false);
      s0 = __builtin_amdgcn_wmma_f32_16x16x32_bf16(false, qa1, false, load16(kr0 + 32 + half * 16), (short)0, s0, false, false);
      s1 = __builtin_amdgcn_wmma_f32_16x16x32_bf16(false, qa0, false, load16(kr1 + half * 16),      (short)0, s1, false, false);
      s1 = __builtin_amdgcn_wmma_f32_16x16x32_bf16(false, qa1, false, load16(kr1 + 32 + half * 16), (short)0, s1, false, false);
    }

    const float mk0 = mb[kt + c];
    const float mk1 = mb[kt + 16 + c];

    // --- online softmax; row r = j + 8*half lives on 16 lanes of each half ---
#pragma unroll
    for (int j = 0; j < 8; ++j) {
      float v0 = s0[j] * scale + mk0;
      float v1 = s1[j] * scale + mk1;
      float mx = fmaxf(v0, v1);
      mx = fmaxf(mx, __shfl_xor(mx, 1, 32));
      mx = fmaxf(mx, __shfl_xor(mx, 2, 32));
      mx = fmaxf(mx, __shfl_xor(mx, 4, 32));
      mx = fmaxf(mx, __shfl_xor(mx, 8, 32));
      float mnew  = fmaxf(mrow[j], mx);
      float alpha = __expf(mrow[j] - mnew);
      float p0 = __expf(v0 - mnew);
      float p1 = __expf(v1 - mnew);
      float rs = p0 + p1;
      rs += __shfl_xor(rs, 1, 32);
      rs += __shfl_xor(rs, 2, 32);
      rs += __shfl_xor(rs, 4, 32);
      rs += __shfl_xor(rs, 8, 32);
      lrow[j] = lrow[j] * alpha + rs;
      mrow[j] = mnew;
#pragma unroll
      for (int nt = 0; nt < 4; ++nt) acc[nt][j] *= alpha;
      pl[wid][j + 8 * half][c]      = to_bf16(p0);
      pl[wid][j + 8 * half][16 + c] = to_bf16(p1);
    }

    // --- P (16x32) @ V (32x64), operands from LDS ---
    const bf16_t* prow = &pl[wid][c][0];
    v16bf pa = combine8(*(const v8bf*)(prow + half * 16),
                        *(const v8bf*)(prow + half * 16 + 8));
#pragma unroll
    for (int nt = 0; nt < 4; ++nt) {
      v16bf vb = load16(&ldsV[buf][nt * 16 + c][half * 16]);
      acc[nt] = __builtin_amdgcn_wmma_f32_16x16x32_bf16(false, pa, false, vb, (short)0, acc[nt], false, false);
    }

    __syncthreads();   // protect buf (rewritten by next iteration's stage)
  }

  // --- epilogue: normalize and store [B,S,D] fp32 ---
#pragma unroll
  for (int j = 0; j < 8; ++j) {
    const float invl = 1.0f / lrow[j];
    float* drow = out + ((size_t)b * SS + q0 + j + 8 * half) * DD + h * HD;
#pragma unroll
    for (int nt = 0; nt < 4; ++nt)
      drow[nt * 16 + c] = acc[nt][j] * invl;
  }
}

// ---------------------------------------------------------------------------
extern "C" void kernel_launch(void* const* d_in, const int* in_sizes, int n_in,
                              void* d_out, int out_size, void* d_ws, size_t ws_size,
                              hipStream_t stream) {
  const float* X    = (const float*)d_in[0];  // [B,S,D]
  const float* mask = (const float*)d_in[1];  // [B,1,1,S]
  const float* Wq   = (const float*)d_in[2];
  const float* bq   = (const float*)d_in[3];
  const float* Wk   = (const float*)d_in[4];
  const float* bk   = (const float*)d_in[5];
  const float* Wv   = (const float*)d_in[6];
  const float* bv   = (const float*)d_in[7];
  float* out = (float*)d_out;

  const size_t XN = (size_t)BB * SS * DD;   // 8388608
  const size_t WN = (size_t)DD * DD;        // 1048576

  bf16_t* Xbf  = (bf16_t*)d_ws;
  bf16_t* Wqbf = Xbf  + XN;
  bf16_t* Wkbf = Wqbf + WN;
  bf16_t* Wvbf = Wkbf + WN;
  bf16_t* Qbf  = Wvbf + WN;
  bf16_t* Kbf  = Qbf  + XN;
  bf16_t* Vtbf = Kbf  + XN;     // [B,H,HD,S]

  const int T = 256;
  cvt_f32_bf16_kernel<<<(int)(XN / 4 / T), T, 0, stream>>>((const float4*)X,  (v4bf*)Xbf,  (int)(XN / 4));
  cvt_f32_bf16_kernel<<<(int)(WN / 4 / T), T, 0, stream>>>((const float4*)Wq, (v4bf*)Wqbf, (int)(WN / 4));
  cvt_f32_bf16_kernel<<<(int)(WN / 4 / T), T, 0, stream>>>((const float4*)Wk, (v4bf*)Wkbf, (int)(WN / 4));
  cvt_f32_bf16_kernel<<<(int)(WN / 4 / T), T, 0, stream>>>((const float4*)Wv, (v4bf*)Wvbf, (int)(WN / 4));

  dim3 gg(BB * SS / 32, DD / 64);   // (256, 16)
  qkv_gemm_kernel<<<gg, 32, 0, stream>>>(Xbf, Wqbf, bq, Qbf, 0);
  qkv_gemm_kernel<<<gg, 32, 0, stream>>>(Xbf, Wkbf, bk, Kbf, 0);
  qkv_gemm_kernel<<<gg, 32, 0, stream>>>(Xbf, Wvbf, bv, Vtbf, 1);

  dim3 ga(SS / 16 / 4, BB * HH);    // (32, 64)
  flash_attn_kernel<<<ga, 128, 0, stream>>>(Qbf, Kbf, Vtbf, mask, out);
}
